// EdgeContrastiveLoss_5265629905503
// MI455X (gfx1250) — matrix operand order repaired and checked by reading, not compiled, not run on hardware
//
#include <hip/hip_runtime.h>
#include <cstdint>

// ---------------------------------------------------------------------------
// EdgeContrastiveLoss for MI455X (gfx1250, wave32, WMMA)
// Sparse formulation: only the 8 positives + 255 sampled negatives per dst row
// are computed, batched 16-dst-at-a-time through v_wmma_f32_16x16x32_bf16 and
// extracted from the 16x16 product diagonal. Inner loop is software-pipelined:
// next tile's gathered src row is in flight while the current WMMA executes.
// ---------------------------------------------------------------------------

typedef __attribute__((ext_vector_type(16))) __bf16        v16bf;
typedef __attribute__((ext_vector_type(8)))  float         v8f;
typedef __attribute__((ext_vector_type(4)))  unsigned int  uintx4;

struct Pack32 { uintx4 lo; uintx4 hi; };
static_assert(sizeof(Pack32) == 32, "Pack32 must be 32 bytes");

#define CHANNELS 64
#define N_SRC    8192
#define N_DST    8192
#define K_POS    8
#define MAX_NEG  255
#define N_EDGE   (N_DST * K_POS)     // 65536
#define N_CAND   (N_DST * MAX_NEG)   // 2088960
#define N_GROUPS (N_DST / 16)        // 512

// ---------------------------------------------------------------------------
// helpers
// ---------------------------------------------------------------------------
__device__ __forceinline__ uint32_t rotl32(uint32_t x, int d) {
  return (x << d) | (x >> (32 - d));
}

// Threefry-2x32-20 exactly as JAX implements it (key = [0, 42] for seed 42).
__device__ __forceinline__ void threefry2x32(uint32_t ks0, uint32_t ks1,
                                             uint32_t x0, uint32_t x1,
                                             uint32_t* o0, uint32_t* o1) {
  const uint32_t ks2 = ks0 ^ ks1 ^ 0x1BD11BDAu;
  x0 += ks0; x1 += ks1;
#define TF_R(r) { x0 += x1; x1 = rotl32(x1, r); x1 ^= x0; }
  TF_R(13) TF_R(15) TF_R(26) TF_R(6)
  x0 += ks1; x1 += ks2 + 1u;
  TF_R(17) TF_R(29) TF_R(16) TF_R(24)
  x0 += ks2; x1 += ks0 + 2u;
  TF_R(13) TF_R(15) TF_R(26) TF_R(6)
  x0 += ks0; x1 += ks1 + 3u;
  TF_R(17) TF_R(29) TF_R(16) TF_R(24)
  x0 += ks1; x1 += ks2 + 4u;
  TF_R(13) TF_R(15) TF_R(26) TF_R(6)
  x0 += ks2; x1 += ks0 + 5u;
#undef TF_R
  *o0 = x0; *o1 = x1;
}

__device__ __forceinline__ uint16_t f32_to_bf16(float f) {
  uint32_t u = __builtin_bit_cast(uint32_t, f);
  uint32_t lsb = (u >> 16) & 1u;
  u += 0x7FFFu + lsb;                 // round-to-nearest-even
  return (uint16_t)(u >> 16);
}

__device__ __forceinline__ v16bf load_bf16x16(const uint16_t* p0,
                                              const uint16_t* p1) {
  Pack32 pk;
  pk.lo = *(const uintx4*)p0;   // 16B chunk
  pk.hi = *(const uintx4*)p1;   // 16B chunk
  return __builtin_bit_cast(v16bf, pk);
}

// Diagonal element D[M][M]: lanes 0-7 hold it in VGPR=lane, lanes 24-31 in
// VGPR=lane-24; unrolled v_cndmask chain extracts acc[vsel].
__device__ __forceinline__ float diag_extract(v8f acc, int vsel) {
  float sim = acc[0];
#pragma unroll
  for (int i = 1; i < 8; ++i) sim = (vsel == i) ? acc[i] : sim;
  return sim;
}

// ---------------------------------------------------------------------------
// prep kernels
// ---------------------------------------------------------------------------
__global__ void bf16_convert_kernel(const float* __restrict__ x,
                                    uint16_t* __restrict__ y, int n) {
  int i = blockIdx.x * blockDim.x + threadIdx.x;
  if (i < n) y[i] = f32_to_bf16(x[i]);
}

// H = dst_x @ W  (f32 accumulate, bf16 store). 8192x64x64: negligible FLOPs.
__global__ void h_kernel(const float* __restrict__ dst_x,
                         const float* __restrict__ W,
                         uint16_t* __restrict__ ht) {
  int tid = blockIdx.x * blockDim.x + threadIdx.x;
  if (tid >= N_DST * CHANNELS) return;
  int row = tid >> 6, col = tid & 63;
  const float* x = dst_x + (size_t)row * CHANNELS;
  float acc = 0.0f;
#pragma unroll
  for (int k = 0; k < CHANNELS; ++k) acc = fmaf(x[k], W[k * CHANNELS + col], acc);
  ht[tid] = f32_to_bf16(acc);
}

// Candidate table: exact JAX randint replication + validity vs the 8 positives.
// Stored as u16: bits[12:0]=src index, bit15 = valid (not a positive edge).
__global__ void cand_kernel(const int* __restrict__ edge_src,
                            uint16_t* __restrict__ cand) {
  int tid = blockIdx.x * blockDim.x + threadIdx.x;
  if (tid >= N_CAND) return;
  const uint32_t half = (uint32_t)N_CAND / 2u;
  uint32_t x0, x1, r0, r1;
  bool use0 = ((uint32_t)tid < half);
  if (use0) { x0 = (uint32_t)tid;        x1 = (uint32_t)tid + half; }
  else      { x0 = (uint32_t)tid - half; x1 = (uint32_t)tid;        }
  threefry2x32(0u, 42u, x0, x1, &r0, &r1);
  uint32_t bits = use0 ? r0 : r1;
  // span 8192 is a power of two <= 2^16 -> randint == bits & 0x1FFF
  uint32_t c = bits & 8191u;
  int d = tid / MAX_NEG;
  const int* pos = edge_src + d * K_POS;
  bool hit = false;
#pragma unroll
  for (int k = 0; k < K_POS; ++k) hit = hit || ((uint32_t)pos[k] == c);
  cand[tid] = (uint16_t)(c | (hit ? 0u : 0x8000u));
}

// ---------------------------------------------------------------------------
// main kernel: one block = 16 dst rows, 8 waves. Wave w handles positive tile
// w (peeled) plus candidate tiles j = w, w+8, ... (software-pipelined).
// ---------------------------------------------------------------------------
__global__ __launch_bounds__(256) void sim_kernel(
    const int* __restrict__ edge_src,
    const uint16_t* __restrict__ srcb,   // src_x in bf16, row-major [8192][64]
    const uint16_t* __restrict__ htb,    // H = dst_x@W in bf16 [8192][64]
    const uint16_t* __restrict__ cand,   // candidate table [8192][255]
    float* __restrict__ partial) {
  __shared__ uintx4 s_cand4[(16 * MAX_NEG) / 8];   // 16x255 u16 = 8160 B
  __shared__ float s_negw[8][16];
  __shared__ float s_cntw[8][16];
  __shared__ float s_neg[16];
  __shared__ float s_cnt[16];
  __shared__ float s_pos[128];

  const int tid  = threadIdx.x;
  const int wave = tid >> 5;
  const int lane = tid & 31;
  const int lo   = lane & 15;
  const int hi   = lane >> 4;
  const int base = blockIdx.x << 4;

  // Stage this block's candidate slice into LDS with coalesced b128 loads.
  {
    const uintx4* g = (const uintx4*)(cand + (size_t)base * MAX_NEG);
#pragma unroll
    for (int i = 0; i < 2; ++i) {
      int idx = tid + i * 256;
      if (idx < (16 * MAX_NEG) / 8) s_cand4[idx] = g[idx];
    }
  }
  const uint16_t* s_cand = (const uint16_t*)s_cand4;

  // B operand: column N=lo holds H row (base+lo); lane-half h takes K chunks
  // [8h,8h+8) and [16+8h,24+8h), matching the A operand order.
  const uint16_t* hrow = htb + (size_t)(base + lo) * CHANNELS;
  const v16bf B0 = load_bf16x16(hrow + 8 * hi,      hrow + 16 + 8 * hi);
  const v16bf B1 = load_bf16x16(hrow + 32 + 8 * hi, hrow + 48 + 8 * hi);

  const bool isdiag = (lane < 8) || (lane >= 24);
  const int  M      = (lane < 8) ? lane : (lane - 16);  // diag row this lane owns
  const int  vsel   = lane & 7;

  // ---- peeled positive tile: wave w computes positives k = w of all 16 dst
  {
    int arow = edge_src[(base + lo) * K_POS + wave];
    const uint16_t* ap = srcb + (size_t)arow * CHANNELS;
    v16bf A0 = load_bf16x16(ap + 8 * hi,      ap + 16 + 8 * hi);
    v16bf A1 = load_bf16x16(ap + 32 + 8 * hi, ap + 48 + 8 * hi);
    v8f acc = {};
    acc = __builtin_amdgcn_wmma_f32_16x16x32_bf16(false, A0, false, B0,
                                                  (short)0, acc, false, false);
    acc = __builtin_amdgcn_wmma_f32_16x16x32_bf16(false, A1, false, B1,
                                                  (short)0, acc, false, false);
    float e = __expf(diag_extract(acc, vsel) * 10.0f);  // exp(sim/TEMP)
    if (isdiag) s_pos[M * K_POS + wave] = e;
  }

  __syncthreads();   // s_cand ready (also covers s_pos ordering later)

  // ---- candidate tiles, fixed 32-iteration uniform loop, 1-deep pipeline
  const uint16_t* crow = s_cand + lo * MAX_NEG;
  float negs = 0.0f, cnts = 0.0f;

  uint32_t cv = crow[wave];
  {
    const uint16_t* ap = srcb + (size_t)(cv & 8191u) * CHANNELS;
    // current-tile A operand (consumed next)
  }
  const uint16_t* ap0 = srcb + (size_t)(cv & 8191u) * CHANNELS;
  v16bf A0 = load_bf16x16(ap0 + 8 * hi,      ap0 + 16 + 8 * hi);
  v16bf A1 = load_bf16x16(ap0 + 32 + 8 * hi, ap0 + 48 + 8 * hi);

#pragma unroll 1
  for (int k = 0; k < 32; ++k) {
    const int j = wave + (k << 3);                 // current tile (uniform/wave)
    int jn = wave + ((k + 1) << 3);                // prefetch tile, clamped
    jn = (jn > MAX_NEG - 1) ? (MAX_NEG - 1) : jn;

    // prefetch next tile's index + gathered src row (overlaps current WMMA)
    uint32_t cvn = crow[jn];
    const uint16_t* apn = srcb + (size_t)(cvn & 8191u) * CHANNELS;
    v16bf N0 = load_bf16x16(apn + 8 * hi,      apn + 16 + 8 * hi);
    v16bf N1 = load_bf16x16(apn + 32 + 8 * hi, apn + 48 + 8 * hi);

    v8f acc = {};
    acc = __builtin_amdgcn_wmma_f32_16x16x32_bf16(false, A0, false, B0,
                                                  (short)0, acc, false, false);
    acc = __builtin_amdgcn_wmma_f32_16x16x32_bf16(false, A1, false, B1,
                                                  (short)0, acc, false, false);

    float e = __expf(diag_extract(acc, vsel) * 10.0f);
    // valid iff not a positive (bit15) and tile not the padded over-run
    float val = ((cv & 0x8000u) && (j < MAX_NEG)) ? 1.0f : 0.0f;
    negs += val * e;
    cnts += val;

    cv = cvn; A0 = N0; A1 = N1;
  }

  if (isdiag) { s_negw[wave][M] = negs; s_cntw[wave][M] = cnts; }
  __syncthreads();

  // deterministic fixed-order cross-wave reduction (no float atomics)
  if (tid < 16) {
    float sn = 0.0f, sc = 0.0f;
#pragma unroll
    for (int w = 0; w < 8; ++w) { sn += s_negw[w][tid]; sc += s_cntw[w][tid]; }
    s_neg[tid] = sn; s_cnt[tid] = sc;
  }
  __syncthreads();

  if (tid < 128) {
    int m = tid >> 3;
    float p = s_pos[tid];
    // -log(p/(p+neg)) / log(cnt+1)
    float per = __logf((p + s_neg[m]) / p) / __logf(s_cnt[m] + 1.0f);
    s_pos[tid] = per;
  }
  __syncthreads();

  if (tid == 0) {
    float s = 0.0f;
    for (int i = 0; i < 128; ++i) s += s_pos[i];
    partial[blockIdx.x] = s * (1.0f / (float)N_EDGE);
  }
}

// deterministic final reduce over the 512 block partials
__global__ void reduce_kernel(const float* __restrict__ partial,
                              float* __restrict__ out) {
  __shared__ float sh[256];
  int tid = threadIdx.x;
  sh[tid] = partial[tid] + partial[tid + 256];
  __syncthreads();
  for (int off = 128; off > 0; off >>= 1) {
    if (tid < off) sh[tid] += sh[tid + off];
    __syncthreads();
  }
  if (tid == 0) out[0] = sh[0];
}

// ---------------------------------------------------------------------------
// launch
// ---------------------------------------------------------------------------
extern "C" void kernel_launch(void* const* d_in, const int* in_sizes, int n_in,
                              void* d_out, int out_size, void* d_ws,
                              size_t ws_size, hipStream_t stream) {
  (void)in_sizes; (void)n_in; (void)out_size; (void)ws_size;

  const float* src_x    = (const float*)d_in[0];
  const float* dst_x    = (const float*)d_in[1];
  const float* W        = (const float*)d_in[2];
  const int*   edge     = (const int*)d_in[3];
  const int*   edge_src = edge;  // edge_index[0] = first N_EDGE ints

  char* ws = (char*)d_ws;
  uint16_t* srcb    = (uint16_t*)(ws);                 // 1,048,576 B
  uint16_t* htb     = (uint16_t*)(ws + 1048576);       // 1,048,576 B
  uint16_t* cand    = (uint16_t*)(ws + 2097152);       // 4,177,920 B
  float*    partial = (float*)   (ws + 6275072);       //     2,048 B

  bf16_convert_kernel<<<(N_SRC * CHANNELS + 255) / 256, 256, 0, stream>>>(
      src_x, srcb, N_SRC * CHANNELS);
  h_kernel<<<(N_DST * CHANNELS + 255) / 256, 256, 0, stream>>>(dst_x, W, htb);
  cand_kernel<<<(N_CAND + 255) / 256, 256, 0, stream>>>(edge_src, cand);
  sim_kernel<<<N_GROUPS, 256, 0, stream>>>(edge_src, srcb, htb, cand, partial);
  reduce_kernel<<<1, 256, 0, stream>>>(partial, (float*)d_out);
}
